// DeepSigNet_28389733827077
// MI455X (gfx1250) — matrix operand order repaired and checked by reading, not compiled, not run on hardware
//
#include <hip/hip_runtime.h>

// ---------------------------------------------------------------------------
// DeepSigNet on gfx1250 (MI455X, wave32, WMMA).
// s3 never materialized: its per-step increment is rank-structured, so
// W1_3 . s3[l] = prefix-sum over l of  Delta3 = W1_3 x G  (G on-the-fly),
// computed with v_wmma_f32_16x16x32_bf16 (f32 accumulate), N = all 2048 (b,l)
// columns in parallel; only the tiny (s1,s2) recursion stays sequential.
// Operand tables are replica-padded so every inner loop is branch-free and
// the B-fragment build uses contiguous (vectorizable) loads.
// ---------------------------------------------------------------------------

typedef __attribute__((ext_vector_type(16))) __bf16 v16bf;
typedef __attribute__((ext_vector_type(8)))  float  v8f;
typedef __attribute__((ext_vector_type(4), aligned(4))) float f32x4u;  // align-4 quad

#define C1     34
#define C1SQ   1156            // 34*34
#define S3LEN  39304           // 34^3
#define NKT3   1229            // ceil(39304/32); last tile peeled
#define NKTS   39              // small K axis: [n1 -> 64 | n2 -> 1220] pad 1248
#define NCOL   2048            // B*L = 8*256
#define S1TOT  40494           // W1 row length

#define DXSTR  52              // dx:  34 + 16 head-replica + pad
#define S1STR  36              // s1:  34 + 1 zero pad
#define N1STR  64              // n1:  34 + zero pad (tile-aligned boundary)
#define B2STR  1184            // b2:  1156 + 16 head-replica + pad
#define E2STR  1160            // e2:  1156 + 1 zero pad
#define N2STR  1184            // n2:  1156 + 28 zero pad

// workspace layout (float offsets; bf16 regions addressed as ushort)
#define OFF_DX   0u            // [2048][52]
#define OFF_S1P  106496u       // [2048][36]
#define OFF_N1   180224u       // [2048][64]
#define OFF_B2   311296u       // [2048][1184]
#define OFF_E2   2736128u      // [2048][1160]
#define OFF_N2   5111808u      // [2048][1184]
#define OFF_HS   7536640u      // [2048][16]
#define OFF_D3   7569408u      // [2048][16]
#define OFF_A3   7602176u      // bf16 [1229][32][16] packed W1_3 A-fragments
#define OFF_AS   7916800u      // bf16 [  39][32][16] packed W1_{1,2} A-fragments
// total 7,926,784 floats ~= 31.7 MB

// ---------------------------------------------------------------------------
// K1: sequential recursion on (s1, s2) only; emits per-(b,l) operand tables
// (with replica/zero pads). One block per batch element, state in LDS.
// ---------------------------------------------------------------------------
__global__ __launch_bounds__(128) void k1_scan(const float* __restrict__ x,
                                               float* __restrict__ ws) {
    const int b = blockIdx.x;
    const int t = threadIdx.x;
    __shared__ float s1s[C1];
    __shared__ float s2s[C1SQ];
    __shared__ float dxs[C1];
    if (t < C1) s1s[t] = 0.f;
    for (int i = t; i < C1SQ; i += 128) s2s[i] = 0.f;
    __syncthreads();

    float* dxW  = ws + OFF_DX;
    float* s1pW = ws + OFF_S1P;
    float* n1W  = ws + OFF_N1;
    float* b2W  = ws + OFF_B2;
    float* e2W  = ws + OFF_E2;
    float* n2W  = ws + OFF_N2;

    for (int l = 0; l < 256; ++l) {
        const int col = b * 256 + l;
        if (t < C1) {
            float cur = (t < 33) ? x[(b * 256 + l) * 33 + t] : (float)l * (1.0f / 255.0f);
            float prv = 0.f;
            if (l > 0)
                prv = (t < 33) ? x[(b * 256 + l - 1) * 33 + t] : (float)(l - 1) * (1.0f / 255.0f);
            float d = cur - prv;
            dxs[t] = d;
            dxW [col * DXSTR + t] = d;
            s1pW[col * S1STR + t] = s1s[t];
            n1W [col * N1STR + t] = s1s[t] + d;
        } else if (t < N1STR) {
            n1W[col * N1STR + t] = 0.f;                       // n1 zero pad
        }
        if (t >= 64 && t < 64 + (N2STR - C1SQ))
            n2W[col * N2STR + C1SQ + (t - 64)] = 0.f;         // n2 zero tail
        __syncthreads();
        // pads that need dxs populated
        if (t < 16) {
            dxW[col * DXSTR + C1 + t]   = dxs[t];             // dx head replica
            b2W[col * B2STR + C1SQ + t] = 0.5f * dxs[0] * dxs[t]; // b2 head replica
        }
        if (t == 16) e2W [col * E2STR + C1SQ] = 0.f;          // e2 pad
        if (t == 17) s1pW[col * S1STR + C1]   = 0.f;          // s1 pad
        for (int idx = t; idx < C1SQ; idx += 128) {
            int i = idx / C1, j = idx - i * C1;
            float b2v = 0.5f * dxs[i] * dxs[j];
            float s2v = s2s[idx];
            float e2v = s2v + b2v * (1.0f / 3.0f);            // s2 + b2/3
            float n2v = s2v + s1s[i] * dxs[j] + b2v;          // Chen update
            b2W[col * B2STR + idx] = b2v;
            e2W[col * E2STR + idx] = e2v;
            n2W[col * N2STR + idx] = n2v;
            s2s[idx] = n2v;
        }
        __syncthreads();
        if (t < C1) s1s[t] += dxs[t];
        __syncthreads();
    }
}

// ---------------------------------------------------------------------------
// K2: pack W1 into bf16 A-fragments as 16x32 WMMA A tiles in this file's own k order (not AMD's table order):
// lane L (M=L%16, g=L/16), VGPR pair p: Kloc = 16g+2p+q (contiguous, matches B).
// Small part uses remapped K axis: [0,34)->W1_1, [64,1220)->W1_2, else 0.
// ---------------------------------------------------------------------------
__global__ __launch_bounds__(32) void k2_pack(const float* __restrict__ W1,
                                              unsigned short* __restrict__ A3,
                                              unsigned short* __restrict__ AS) {
    const int kt = blockIdx.x;
    const int L = threadIdx.x, M = L & 15, g = L >> 4;
    const bool sm = (kt >= NKT3);
    const int ktl = sm ? kt - NKT3 : kt;
    unsigned short* dst = (sm ? AS : A3) + (ktl * 32 + L) * 16;

    __bf16 hh[16];
#pragma unroll
    for (int p = 0; p < 8; ++p) {
#pragma unroll
        for (int q = 0; q < 2; ++q) {
            int Kloc = 16 * g + 2 * p + q;
            int K = ktl * 32 + Kloc;
            float v = 0.f;
            if (M < 8) {
                if (sm) {
                    if (K < C1)                        v = W1[M * S1TOT + K];
                    else if (K >= 64 && K < 64 + C1SQ) v = W1[M * S1TOT + C1 + (K - 64)];
                } else {
                    if (K < S3LEN)                     v = W1[M * S1TOT + C1 + C1SQ + K];
                }
            }
            hh[2 * p + q] = (__bf16)v;
        }
    }
#pragma unroll
    for (int u = 0; u < 8; ++u)
        ((unsigned int*)dst)[u] = (unsigned int)__builtin_bit_cast(unsigned short, hh[2 * u]) |
                                  ((unsigned int)__builtin_bit_cast(unsigned short, hh[2 * u + 1]) << 16);
}

// ---------------------------------------------------------------------------
// K3: the WMMA GEMMs, one wave per 16-column tile. Branch-free inner loops;
// replica padding makes dx/b2 reads contiguous, e2/s1 take <=2 values per run.
// B 32x16 layout: lane L holds N=L%16, K = kt*32 + 16g + (0..15) contiguous.
// ---------------------------------------------------------------------------
union BFr { v16bf v; __bf16 e[16]; };

__global__ __launch_bounds__(32) void k3_gemm(const unsigned short* __restrict__ A3,
                                              const unsigned short* __restrict__ AS,
                                              float* __restrict__ ws) {
    const int tile = blockIdx.x;               // 0..127
    const int L = threadIdx.x;
    const int g = L >> 4, n = L & 15;
    const int col = tile * 16 + n;

    const float* dxp = ws + OFF_DX  + col * DXSTR;
    const float* s1p = ws + OFF_S1P + col * S1STR;
    const float* n1c = ws + OFF_N1  + col * N1STR;
    const float* b2p = ws + OFF_B2  + col * B2STR;
    const float* e2p = ws + OFF_E2  + col * E2STR;
    const float* n2c = ws + OFF_N2  + col * N2STR;
    float* hsW = ws + OFF_HS;
    float* d3W = ws + OFF_D3;

    // ---- small GEMM: W1_{1,2} . [n1; n2] (K boundary tile-aligned at 64) ----
    v8f acc = {};
#pragma unroll
    for (int kt = 0; kt < 2; ++kt) {           // K in [0,64): padded n1 only
        v16bf a = *(const v16bf*)(const void*)(AS + (kt * 32 + L) * 16);
        const int base = kt * 32 + 16 * g;
        BFr bf;
#pragma unroll
        for (int t = 0; t < 16; ++t) bf.e[t] = (__bf16)n1c[base + t];
        acc = __builtin_amdgcn_wmma_f32_16x16x32_bf16(false, a, false, bf.v,
                                                      (short)0, acc, false, false);
    }
    for (int kt = 2; kt < NKTS; ++kt) {        // K in [64,1248): padded n2 only
        v16bf a = *(const v16bf*)(const void*)(AS + (kt * 32 + L) * 16);
        const int base = kt * 32 + 16 * g - 64;
        BFr bf;
#pragma unroll
        for (int t = 0; t < 16; ++t) bf.e[t] = (__bf16)n2c[base + t];
        acc = __builtin_amdgcn_wmma_f32_16x16x32_bf16(false, a, false, bf.v,
                                                      (short)0, acc, false, false);
    }
#pragma unroll
    for (int r = 0; r < 8; ++r) hsW[col * 16 + 8 * g + r] = acc[r];

    // ---- main GEMM: W1_3 . g,  g[ijk] = e2[ij]*dx[k] + s1[i]*b2[jk] ----
    // Run-start indices for K = base: k0 = K%34, ij0 = K/34, j0 = ij0%34,
    // i0 = ij0/34, r0 = K - i0*1156 (b2 offset, replica-padded across wraps).
    v8f acc3 = {};
    int k0 = (16 * g) % C1;
    int ij0 = (16 * g) / C1;
    int j0 = ij0, i0 = 0, r0 = 16 * g;
#pragma unroll 2
    for (int kt = 0; kt < NKT3 - 1; ++kt) {    // K < 39296: fully in range
        v16bf a = *(const v16bf*)(const void*)(A3 + (kt * 32 + L) * 16);
        __builtin_prefetch((const void*)(A3 + ((kt + 1) * 32 + L) * 16), 0, 1);
        f32x4u dxv0 = *(const f32x4u*)(dxp + k0);
        f32x4u dxv1 = *(const f32x4u*)(dxp + k0 + 4);
        f32x4u dxv2 = *(const f32x4u*)(dxp + k0 + 8);
        f32x4u dxv3 = *(const f32x4u*)(dxp + k0 + 12);
        f32x4u b2v0 = *(const f32x4u*)(b2p + r0);
        f32x4u b2v1 = *(const f32x4u*)(b2p + r0 + 4);
        f32x4u b2v2 = *(const f32x4u*)(b2p + r0 + 8);
        f32x4u b2v3 = *(const f32x4u*)(b2p + r0 + 12);
        float e2a = e2p[ij0], e2b = e2p[ij0 + 1];
        float s1a = s1p[i0],  s1b = s1p[i0 + 1];
        float s1w = (j0 == C1 - 1) ? s1b : s1a;
        const int twrap = C1 - k0;             // first t where ij/i step
        float dxe[16], b2e[16];
#pragma unroll
        for (int u = 0; u < 4; ++u) {
            dxe[u]      = dxv0[u]; dxe[4 + u]  = dxv1[u];
            dxe[8 + u]  = dxv2[u]; dxe[12 + u] = dxv3[u];
            b2e[u]      = b2v0[u]; b2e[4 + u]  = b2v1[u];
            b2e[8 + u]  = b2v2[u]; b2e[12 + u] = b2v3[u];
        }
        BFr bf;
#pragma unroll
        for (int t = 0; t < 16; ++t) {
            bool w = (t >= twrap);
            float e2v = w ? e2b : e2a;
            float s1v = w ? s1w : s1a;
            bf.e[t] = (__bf16)(e2v * dxe[t] + s1v * b2e[t]);
        }
        acc3 = __builtin_amdgcn_wmma_f32_16x16x32_bf16(false, a, false, bf.v,
                                                       (short)0, acc3, false, false);
        // advance base by 32
        k0 += 32;
        r0 += 32;
        if (k0 >= C1) {
            k0 -= C1;
            ++ij0; ++j0;
            if (j0 == C1) { j0 = 0; ++i0; r0 -= C1SQ; }
        }
    }
    {                                          // peeled last ktile (kt = 1228)
        const int kt = NKT3 - 1;
        v16bf a = *(const v16bf*)(const void*)(A3 + (kt * 32 + L) * 16);
        const int base = kt * 32 + 16 * g;
        int pi = base / C1SQ;
        int pr = base - pi * C1SQ;
        int pij = base / C1;
        int pj = pij - pi * C1;
        int pk = base - pij * C1;
        f32x4u dxv0 = *(const f32x4u*)(dxp + pk);
        f32x4u dxv1 = *(const f32x4u*)(dxp + pk + 4);
        f32x4u dxv2 = *(const f32x4u*)(dxp + pk + 8);
        f32x4u dxv3 = *(const f32x4u*)(dxp + pk + 12);
        f32x4u b2v0 = *(const f32x4u*)(b2p + pr);
        f32x4u b2v1 = *(const f32x4u*)(b2p + pr + 4);
        f32x4u b2v2 = *(const f32x4u*)(b2p + pr + 8);
        f32x4u b2v3 = *(const f32x4u*)(b2p + pr + 12);
        float e2a = e2p[pij], e2b = e2p[pij + 1];
        float s1a = s1p[pi],  s1b = s1p[pi + 1];
        float s1w = (pj == C1 - 1) ? s1b : s1a;
        const int twrap = C1 - pk;
        float dxe[16], b2e[16];
#pragma unroll
        for (int u = 0; u < 4; ++u) {
            dxe[u]      = dxv0[u]; dxe[4 + u]  = dxv1[u];
            dxe[8 + u]  = dxv2[u]; dxe[12 + u] = dxv3[u];
            b2e[u]      = b2v0[u]; b2e[4 + u]  = b2v1[u];
            b2e[8 + u]  = b2v2[u]; b2e[12 + u] = b2v3[u];
        }
        BFr bf;
#pragma unroll
        for (int t = 0; t < 16; ++t) {
            bool w = (t >= twrap);
            float e2v = w ? e2b : e2a;
            float s1v = w ? s1w : s1a;
            float v = e2v * dxe[t] + s1v * b2e[t];
            bf.e[t] = (__bf16)((base + t < S3LEN) ? v : 0.f);
        }
        acc3 = __builtin_amdgcn_wmma_f32_16x16x32_bf16(false, a, false, bf.v,
                                                       (short)0, acc3, false, false);
    }
#pragma unroll
    for (int r = 0; r < 8; ++r) d3W[col * 16 + 8 * g + r] = acc3[r];
}

// ---------------------------------------------------------------------------
// K4: prefix-sum Delta3 over l (=> W1_3 . s3), bias+relu, W2, then the tiny
// C=4 depth-3 signature scan fused in registers (one lane per batch), Wl.
// ---------------------------------------------------------------------------
__global__ void k4_final(const float* __restrict__ ws, const float* __restrict__ b1,
                         const float* __restrict__ W2, const float* __restrict__ bias2,
                         const float* __restrict__ Wl, const float* __restrict__ bl,
                         float* __restrict__ out) {
    const int b = threadIdx.x;
    if (b >= 8) return;
    const float* hsW = ws + OFF_HS;
    const float* d3W = ws + OFF_D3;

    float P[8], S1[4], S2[16], S3[64], cp[4];
#pragma unroll
    for (int h = 0; h < 8; ++h) P[h] = 0.f;
#pragma unroll
    for (int i = 0; i < 4; ++i) { S1[i] = 0.f; cp[i] = 0.f; }
#pragma unroll
    for (int i = 0; i < 16; ++i) S2[i] = 0.f;
#pragma unroll
    for (int i = 0; i < 64; ++i) S3[i] = 0.f;
    const float blv = bl[0];

    for (int l = 0; l < 256; ++l) {
        const int col = b * 256 + l;
        float hv[8];
#pragma unroll
        for (int h = 0; h < 8; ++h) {
            P[h] += d3W[col * 16 + h];                       // inclusive cumsum
            float pre = hsW[col * 16 + h] + b1[h] + P[h];
            hv[h] = pre > 0.f ? pre : 0.f;
        }
        float c[4];
#pragma unroll
        for (int o = 0; o < 4; ++o) {
            float a = bias2[o];
#pragma unroll
            for (int h = 0; h < 8; ++h) a += W2[o * 8 + h] * hv[h];
            c[o] = a;
        }
        float dc[4];
#pragma unroll
        for (int o = 0; o < 4; ++o) { dc[o] = c[o] - cp[o]; cp[o] = c[o]; }

        float bb[16];
#pragma unroll
        for (int i = 0; i < 4; ++i)
#pragma unroll
            for (int j = 0; j < 4; ++j) bb[i * 4 + j] = 0.5f * dc[i] * dc[j];

        float acc = blv;
        float n1v[4];
#pragma unroll
        for (int i = 0; i < 4; ++i) { n1v[i] = S1[i] + dc[i]; acc += Wl[i] * n1v[i]; }
        float n2v[16];
#pragma unroll
        for (int i = 0; i < 4; ++i)
#pragma unroll
            for (int j = 0; j < 4; ++j) {
                int ij = i * 4 + j;
                n2v[ij] = S2[ij] + S1[i] * dc[j] + bb[ij];
                acc += Wl[4 + ij] * n2v[ij];
            }
#pragma unroll
        for (int i = 0; i < 4; ++i)
#pragma unroll
            for (int j = 0; j < 4; ++j)
#pragma unroll
                for (int k = 0; k < 4; ++k) {
                    int ijk = (i * 4 + j) * 4 + k;
                    float v = S3[ijk] + S2[i * 4 + j] * dc[k] + S1[i] * bb[j * 4 + k]
                              + bb[i * 4 + j] * dc[k] * (1.0f / 3.0f);
                    S3[ijk] = v;
                    acc += Wl[20 + ijk] * v;
                }
#pragma unroll
        for (int i = 0; i < 16; ++i) S2[i] = n2v[i];
#pragma unroll
        for (int i = 0; i < 4; ++i) S1[i] = n1v[i];
        out[col] = acc;
    }
}

// ---------------------------------------------------------------------------
extern "C" void kernel_launch(void* const* d_in, const int* in_sizes, int n_in,
                              void* d_out, int out_size, void* d_ws, size_t ws_size,
                              hipStream_t stream) {
    const float* x  = (const float*)d_in[0];
    const float* W1 = (const float*)d_in[1];
    const float* b1 = (const float*)d_in[2];
    const float* W2 = (const float*)d_in[3];
    const float* b2 = (const float*)d_in[4];
    const float* Wl = (const float*)d_in[5];
    const float* bl = (const float*)d_in[6];
    float* ws = (float*)d_ws;
    unsigned short* A3 = (unsigned short*)(ws + OFF_A3);
    unsigned short* AS = (unsigned short*)(ws + OFF_AS);

    hipLaunchKernelGGL(k1_scan, dim3(8), dim3(128), 0, stream, x, ws);
    hipLaunchKernelGGL(k2_pack, dim3(NKT3 + NKTS), dim3(32), 0, stream, W1, A3, AS);
    hipLaunchKernelGGL(k3_gemm, dim3(NCOL / 16), dim3(32), 0, stream, A3, AS, ws);
    hipLaunchKernelGGL(k4_final, dim3(1), dim3(8), 0, stream, ws, b1, W2, b2, Wl, bl,
                       (float*)d_out);
}
